// MultiHeadAttention_1571958030475
// MI455X (gfx1250) — matrix operand order repaired and checked by reading, compiled, not run on hardware
//
#include <hip/hip_runtime.h>
#include <hip/hip_bf16.h>
#include <math.h>

typedef __attribute__((ext_vector_type(16))) __bf16 v16bf;
typedef __attribute__((ext_vector_type(8)))  float  v8f;

#define WMMA_BF16(a, b, c) \
  __builtin_amdgcn_wmma_f32_16x16x32_bf16(false, (a), false, (b), (short)0, (c), false, false)

union BF16x8 { uint4 u; __bf16 h[8]; };

static constexpr int Bsz = 4, Ssz = 2048, Dsz = 1024, Hn = 16, HDsz = 64;
static constexpr int Mrows = Bsz * Ssz;   // 8192

// ---------------------------------------------------------------------------
// Async global->LDS staging (CDNA5 GLOBAL_LOAD_ASYNC_TO_LDS_B128, ASYNCcnt).
// ---------------------------------------------------------------------------
#if defined(__has_builtin)
#if __has_builtin(__builtin_amdgcn_global_load_async_to_lds_b128) && \
    __has_builtin(__builtin_amdgcn_s_wait_asynccnt)
#define USE_ASYNC_LDS 1
#endif
#if __has_builtin(__builtin_amdgcn_update_dpp) && __has_builtin(__builtin_amdgcn_permlane16)
#define USE_VALU_XLANE 1
#endif
#endif
#ifndef USE_ASYNC_LDS
#define USE_ASYNC_LDS 0
#endif
#ifndef USE_VALU_XLANE
#define USE_VALU_XLANE 0
#endif

#if USE_ASYNC_LDS
typedef int v4i_vec __attribute__((vector_size(16)));
typedef __attribute__((address_space(1))) v4i_vec gbl_v4i;
typedef __attribute__((address_space(3))) v4i_vec lds_v4i;
__device__ __forceinline__ void async_copy16(const void* g, void* l) {
  __builtin_amdgcn_global_load_async_to_lds_b128(
      (gbl_v4i*)(void*)g, (lds_v4i*)l, 0, 0);
}
template <int N>
__device__ __forceinline__ void wait_async() {
  __builtin_amdgcn_s_wait_asynccnt(N);   // immediate operand required
}
#else
template <int N>
__device__ __forceinline__ void wait_async() {}
#endif

// ---------------------------------------------------------------------------
// Cross-lane max over 16-lane groups, all on the VALU pipe:
// xor1/xor2 via DPP quad_perm, xor4/xor8 via v_permlane16_b32 nibble selects.
// ---------------------------------------------------------------------------
__device__ __forceinline__ float rowmax16(float x) {
#if USE_VALU_XLANE
  int t;
  t = __builtin_amdgcn_update_dpp(0, __float_as_int(x), 0xB1, 0xF, 0xF, true); // quad_perm(1,0,3,2)
  x = fmaxf(x, __int_as_float(t));
  t = __builtin_amdgcn_update_dpp(0, __float_as_int(x), 0x4E, 0xF, 0xF, true); // quad_perm(2,3,0,1)
  x = fmaxf(x, __int_as_float(t));
  t = __builtin_amdgcn_permlane16(__float_as_int(x), __float_as_int(x),
                                  (int)0x32107654, (int)0xBA98FEDC, false, false); // lane^4
  x = fmaxf(x, __int_as_float(t));
  t = __builtin_amdgcn_permlane16(__float_as_int(x), __float_as_int(x),
                                  (int)0xFEDCBA98, (int)0x76543210, false, false); // lane^8
  x = fmaxf(x, __int_as_float(t));
#else
#pragma unroll
  for (int msk = 1; msk < 16; msk <<= 1)
    x = fmaxf(x, __shfl_xor(x, msk, 32));
#endif
  return x;
}

// ---------------------------------------------------------------------------
// fp32 -> bf16 conversion (row-major copy)
// ---------------------------------------------------------------------------
__global__ __launch_bounds__(256)
void f32_to_bf16_kernel(const float* __restrict__ src, __bf16* __restrict__ dst, int n) {
  int i = blockIdx.x * blockDim.x + threadIdx.x;
  if (i < n) dst[i] = (__bf16)src[i];
}

// ---------------------------------------------------------------------------
// fp32 [R][C] -> bf16 [C][R] tiled transpose (weights become W^T once, so the
// GEMM B-stage is a contiguous async copy with no per-iteration shuffling).
// ---------------------------------------------------------------------------
__global__ __launch_bounds__(256)
void transpose_f32_to_bf16_kernel(const float* __restrict__ src, __bf16* __restrict__ dst,
                                  int R, int C) {
  __shared__ float tile[32][33];
  const int tx = threadIdx.x & 31, ty = threadIdx.x >> 5;   // 32 x 8
  const int r0 = blockIdx.y * 32, c0 = blockIdx.x * 32;
#pragma unroll
  for (int i = 0; i < 32; i += 8)
    tile[ty + i][tx] = src[(size_t)(r0 + ty + i) * C + c0 + tx];
  __syncthreads();
#pragma unroll
  for (int i = 0; i < 32; i += 8)
    dst[(size_t)(c0 + ty + i) * R + r0 + tx] = (__bf16)tile[tx][ty + i];
}

// ---------------------------------------------------------------------------
// Tiled WMMA GEMM: C[M,N] = A[M,K] * Wt^T[K,N]  (Wt given as [N][K] bf16).
// Block tile 128x128, k-step 64, 256 threads = 8 waves, each wave 32x64.
// Double-buffered async global->LDS staging for both A and B tiles.
// OUT_MODE: 0 = f32 row-major, 1 = bf16 row-major, 2 = bf16 col-major (V^T)
// ---------------------------------------------------------------------------
template <int OUT_MODE>
__global__ __launch_bounds__(256)
void gemm_bf16_kernel(const __bf16* __restrict__ A, const __bf16* __restrict__ Wt,
                      void* __restrict__ Cout, int M, int N, int K) {
  __shared__ alignas(16) __bf16 As[2][128][72];   // [buf][m][k]
  __shared__ alignas(16) __bf16 Bt[2][128][72];   // [buf][n][k]

  const int tid  = threadIdx.x;
  const int wave = tid >> 5;
  const int lane = tid & 31;
  const int half = lane >> 4;
  const int l16  = lane & 15;
  const int m0 = blockIdx.y * 128;
  const int n0 = blockIdx.x * 128;
  const int wr = wave & 3;    // 32-row strip within block tile
  const int wc = wave >> 2;   // 64-col strip within block tile

  v8f acc[2][4] = {};

  const int srow = tid >> 1;            // stage: 2 threads/row, 128 rows
  const int scol = (tid & 1) * 32;      // 32 elements (64B) per thread

  auto stage = [&](int buf, int k0) {
    const __bf16* ga = A  + (size_t)(m0 + srow) * K + k0 + scol;
    const __bf16* gb = Wt + (size_t)(n0 + srow) * K + k0 + scol;
#if USE_ASYNC_LDS
#pragma unroll
    for (int c = 0; c < 4; ++c) {
      async_copy16(ga + c * 8, &As[buf][srow][scol + c * 8]);
      async_copy16(gb + c * 8, &Bt[buf][srow][scol + c * 8]);
    }
#else
    const uint4* va = (const uint4*)ga;
    const uint4* vb = (const uint4*)gb;
#pragma unroll
    for (int c = 0; c < 4; ++c) {
      *(uint4*)&As[buf][srow][scol + c * 8] = va[c];
      *(uint4*)&Bt[buf][srow][scol + c * 8] = vb[c];
    }
#endif
  };

  stage(0, 0);
  for (int k0 = 0, it = 0; k0 < K; k0 += 64, ++it) {
    const int cur = it & 1;
    const bool more = (k0 + 64) < K;
    if (more) {
      stage(cur ^ 1, k0 + 64);   // prefetch next tile (async)
      wait_async<8>();           // current tile's 8 copies done, next 8 pending
    } else {
      wait_async<0>();           // drain everything
    }
    __syncthreads();

#pragma unroll
    for (int kk = 0; kk < 2; ++kk) {
      v16bf af[2], bfm[4];
#pragma unroll
      for (int mt = 0; mt < 2; ++mt) {
        int row = wr * 32 + mt * 16 + l16;
        BF16x8 lo, hi;
        lo.u = *(const uint4*)&As[cur][row][kk * 32 + half * 16];
        hi.u = *(const uint4*)&As[cur][row][kk * 32 + half * 16 + 8];
#pragma unroll
        for (int e = 0; e < 8; ++e) { af[mt][e] = lo.h[e]; af[mt][8 + e] = hi.h[e]; }
      }
#pragma unroll
      for (int nt = 0; nt < 4; ++nt) {
        int col = wc * 64 + nt * 16 + l16;
        BF16x8 lo, hi;
        lo.u = *(const uint4*)&Bt[cur][col][kk * 32 + half * 16];
        hi.u = *(const uint4*)&Bt[cur][col][kk * 32 + half * 16 + 8];
#pragma unroll
        for (int e = 0; e < 8; ++e) { bfm[nt][e] = lo.h[e]; bfm[nt][8 + e] = hi.h[e]; }
      }
#pragma unroll
      for (int mt = 0; mt < 2; ++mt)
#pragma unroll
        for (int nt = 0; nt < 4; ++nt)
          acc[mt][nt] = WMMA_BF16(af[mt], bfm[nt], acc[mt][nt]);
    }
    __syncthreads();   // all waves done reading 'cur' before it is restaged
  }

  // ---- store (C layout: row = r + half*8, col = l16)
#pragma unroll
  for (int mt = 0; mt < 2; ++mt)
#pragma unroll
    for (int nt = 0; nt < 4; ++nt)
#pragma unroll
      for (int r = 0; r < 8; ++r) {
        int row = m0 + wr * 32 + mt * 16 + r + half * 8;
        int col = n0 + wc * 64 + nt * 16 + l16;
        if (OUT_MODE == 0)
          ((float*)Cout)[(size_t)row * N + col] = acc[mt][nt][r];
        else if (OUT_MODE == 1)
          ((__bf16*)Cout)[(size_t)row * N + col] = (__bf16)acc[mt][nt][r];
        else  // col-major: Vt[col][row]
          ((__bf16*)Cout)[(size_t)col * M + row] = (__bf16)acc[mt][nt][r];
      }
}

// ---------------------------------------------------------------------------
// Causal flash attention. Q,K: [B*S, D] bf16 (head h at cols h*64..h*64+63).
// V: transposed [D, B*S] bf16 so V B-fragments are contiguous 32B loads.
// One wave per 16-row query tile; 32-key inner tiles; base-2 online softmax.
// Row-max: VALU butterfly (DPP+permlane16). Row-sum: WMMA against ones.
// Only the final key tile intersects the diagonal -> mask-free main loop.
// ---------------------------------------------------------------------------
__global__ __launch_bounds__(128)
void attn_kernel(const __bf16* __restrict__ Q, const __bf16* __restrict__ Kb,
                 const __bf16* __restrict__ Vt, __bf16* __restrict__ Ctx) {
  __shared__ alignas(16) __bf16 Plds[4][16][32];

  const int tid  = threadIdx.x;
  const int wave = tid >> 5;
  const int lane = tid & 31;
  const int half = lane >> 4;
  const int l16  = lane & 15;

  const int qchunks = Ssz / 64;                 // 32
  const int bh = blockIdx.x / qchunks;
  const int qc = blockIdx.x % qchunks;
  const int b  = bh / Hn;
  const int h  = bh % Hn;
  const int q0 = qc * 64 + wave * 16;

  const size_t baseRow = (size_t)b * Ssz;
  const int hoff = h * HDsz;

  // Q fragments for k = 0..31 and 32..63
  v16bf aQ[2];
  {
    const __bf16* qrow = Q + (baseRow + q0 + l16) * Dsz + hoff;
#pragma unroll
    for (int kk = 0; kk < 2; ++kk) {
      BF16x8 lo, hi;
      lo.u = *(const uint4*)(qrow + kk * 32 + half * 16);
      hi.u = *(const uint4*)(qrow + kk * 32 + half * 16 + 8);
#pragma unroll
      for (int e = 0; e < 8; ++e) { aQ[kk][e] = lo.h[e]; aQ[kk][8 + e] = hi.h[e]; }
    }
  }

  // constant all-ones B fragment for row-sum WMMA
  v16bf onesb;
#pragma unroll
  for (int e = 0; e < 16; ++e) onesb[e] = (__bf16)1.0f;

  v8f acc[4] = {};                 // ctx accum, hd groups of 16
  float mrow[8], lrow[8];
#pragma unroll
  for (int r = 0; r < 8; ++r) { mrow[r] = -3.0e38f; lrow[r] = 0.f; }
  const float scale2 = 0.125f * 1.44269504089f;  // (1/sqrt(64)) * log2(e)

  auto tile_step = [&](int j, bool masked) {
    // ---- scores: S = Q * K^T for keys [j, j+31]
    v8f sc0 = {}, sc1 = {};
    {
      const __bf16* krow0 = Kb + (baseRow + j + l16) * Dsz + hoff + half * 16;
      const __bf16* krow1 = Kb + (baseRow + j + 16 + l16) * Dsz + hoff + half * 16;
#pragma unroll
      for (int kk = 0; kk < 2; ++kk) {
        v16bf bK0, bK1;
        BF16x8 a0, a1, c0, c1;
        a0.u = *(const uint4*)(krow0 + kk * 32);
        a1.u = *(const uint4*)(krow0 + kk * 32 + 8);
        c0.u = *(const uint4*)(krow1 + kk * 32);
        c1.u = *(const uint4*)(krow1 + kk * 32 + 8);
#pragma unroll
        for (int e = 0; e < 8; ++e) {
          bK0[e] = a0.h[e]; bK0[8 + e] = a1.h[e];
          bK1[e] = c0.h[e]; bK1[8 + e] = c1.h[e];
        }
        sc0 = WMMA_BF16(aQ[kk], bK0, sc0);
        sc1 = WMMA_BF16(aQ[kk], bK1, sc1);
      }
    }

    // ---- (optional) causal mask + base-2 online softmax max update
    float s0[8], s1[8], alpha[8];
#pragma unroll
    for (int r = 0; r < 8; ++r) {
      float a = sc0[r] * scale2;
      float c = sc1[r] * scale2;
      if (masked) {
        int row = q0 + r + half * 8;
        if (j + l16 > row)      a = -3.0e38f;
        if (j + 16 + l16 > row) c = -3.0e38f;
      }
      s0[r] = a; s1[r] = c;
    }
#pragma unroll
    for (int r = 0; r < 8; ++r) {
      float mx = rowmax16(fmaxf(s0[r], s1[r]));
      float mn = fmaxf(mrow[r], mx);
      alpha[r] = exp2f(mrow[r] - mn);
      mrow[r]  = mn;
      s0[r] = exp2f(s0[r] - mn);
      s1[r] = exp2f(s1[r] - mn);
    }

    // ---- P (C-frag layout) -> LDS -> A-frag layout
#pragma unroll
    for (int r = 0; r < 8; ++r) {
      Plds[wave][r + half * 8][l16]      = (__bf16)s0[r];
      Plds[wave][r + half * 8][16 + l16] = (__bf16)s1[r];
    }
    __builtin_amdgcn_wave_barrier();
    asm volatile("" ::: "memory");

    v16bf aP;
    {
      BF16x8 lo, hi;
      lo.u = *(const uint4*)&Plds[wave][l16][half * 16];
      hi.u = *(const uint4*)&Plds[wave][l16][half * 16 + 8];
#pragma unroll
      for (int e = 0; e < 8; ++e) { aP[e] = lo.h[e]; aP[8 + e] = hi.h[e]; }
    }

    // ---- row sums via WMMA: P x ones -> every lane holds its rows' sums
    {
      v8f zero = {};
      v8f sums = WMMA_BF16(aP, onesb, zero);
#pragma unroll
      for (int r = 0; r < 8; ++r)
        lrow[r] = lrow[r] * alpha[r] + sums[r];
    }

    // ---- rescale accum, then acc += P * V (Vt: contiguous 32B per lane)
#pragma unroll
    for (int f = 0; f < 4; ++f)
#pragma unroll
      for (int r = 0; r < 8; ++r)
        acc[f][r] *= alpha[r];
#pragma unroll
    for (int f = 0; f < 4; ++f) {
      v16bf vbf;
      BF16x8 lo, hi;
      const __bf16* vcol = Vt + (size_t)(hoff + f * 16 + l16) * Mrows
                              + baseRow + j + half * 16;
      lo.u = *(const uint4*)(vcol);
      hi.u = *(const uint4*)(vcol + 8);
#pragma unroll
      for (int e = 0; e < 8; ++e) { vbf[e] = lo.h[e]; vbf[8 + e] = hi.h[e]; }
      acc[f] = WMMA_BF16(aP, vbf, acc[f]);
    }
  };

  const int jmask = (q0 + 15) & ~31;   // only tile that touches the diagonal
  for (int j = 0; j < jmask; j += 32)
    tile_step(j, false);
  tile_step(jmask, true);

  // ---- normalize and store ctx (bf16) back in [B*S, D] layout
#pragma unroll
  for (int f = 0; f < 4; ++f)
#pragma unroll
    for (int r = 0; r < 8; ++r) {
      int row = q0 + r + half * 8;
      float o = acc[f][r] / lrow[r];
      Ctx[(baseRow + row) * Dsz + hoff + f * 16 + l16] = (__bf16)o;
    }
}

// ---------------------------------------------------------------------------
// Launch
// ---------------------------------------------------------------------------
extern "C" void kernel_launch(void* const* d_in, const int* in_sizes, int n_in,
                              void* d_out, int out_size, void* d_ws, size_t ws_size,
                              hipStream_t stream) {
  const float* x  = (const float*)d_in[0];
  const float* Wq = (const float*)d_in[1];
  const float* Wk = (const float*)d_in[2];
  const float* Wv = (const float*)d_in[3];
  const float* Wo = (const float*)d_in[4];

  char* ws = (char*)d_ws;
  size_t off = 0;
  auto alloc = [&](size_t bytes) -> void* {
    void* p = ws + off;
    off += (bytes + 255) & ~(size_t)255;
    return p;
  };

  const size_t wbytes = (size_t)Dsz * Dsz * sizeof(__bf16);
  const size_t mbytes = (size_t)Mrows * Dsz * sizeof(__bf16);
  __bf16* wqt = (__bf16*)alloc(wbytes);   // W^T [N][K] bf16
  __bf16* wkt = (__bf16*)alloc(wbytes);
  __bf16* wvt = (__bf16*)alloc(wbytes);
  __bf16* wot = (__bf16*)alloc(wbytes);
  __bf16* Qb  = (__bf16*)alloc(mbytes);
  __bf16* Kb  = (__bf16*)alloc(mbytes);
  __bf16* Vtb = (__bf16*)alloc(mbytes);   // V stored transposed [D, B*S]
  __bf16* xb  = (__bf16*)alloc(mbytes);
  __bf16* ctx = xb;   // reuse: xb dead after projections (stream-ordered)

  const int nX = Mrows * Dsz;
  f32_to_bf16_kernel<<<(nX + 255) / 256, 256, 0, stream>>>(x, xb, nX);
  dim3 tg(Dsz / 32, Dsz / 32), tb(256);
  transpose_f32_to_bf16_kernel<<<tg, tb, 0, stream>>>(Wq, wqt, Dsz, Dsz);
  transpose_f32_to_bf16_kernel<<<tg, tb, 0, stream>>>(Wk, wkt, Dsz, Dsz);
  transpose_f32_to_bf16_kernel<<<tg, tb, 0, stream>>>(Wv, wvt, Dsz, Dsz);
  transpose_f32_to_bf16_kernel<<<tg, tb, 0, stream>>>(Wo, wot, Dsz, Dsz);

  dim3 gg(Dsz / 128, Mrows / 128), gb(256);
  gemm_bf16_kernel<1><<<gg, gb, 0, stream>>>(xb, wqt, Qb,  Mrows, Dsz, Dsz);
  gemm_bf16_kernel<1><<<gg, gb, 0, stream>>>(xb, wkt, Kb,  Mrows, Dsz, Dsz);
  gemm_bf16_kernel<2><<<gg, gb, 0, stream>>>(xb, wvt, Vtb, Mrows, Dsz, Dsz);

  attn_kernel<<<Bsz * Hn * (Ssz / 64), 128, 0, stream>>>(Qb, Kb, Vtb, ctx);

  gemm_bf16_kernel<0><<<gg, gb, 0, stream>>>(ctx, wot, d_out, Mrows, Dsz, Dsz);
}
